// MyGraphLayer_79216376807650
// MI455X (gfx1250) — matrix operand order repaired and checked by reading, not compiled, not run on hardware
//
#include <hip/hip_runtime.h>
#include <hip/hip_bf16.h>

// Problem constants from the reference
#define B_   4
#define N_   2048
#define D_   128
#define OUT_ 64
#define NT   (N_/16)   // 128 row tiles
#define DT   (D_/16)   // 8 col tiles

typedef __attribute__((ext_vector_type(16))) __bf16        v16bf;
typedef __attribute__((ext_vector_type(8)))  float         v8f;
typedef __attribute__((ext_vector_type(4)))  unsigned int  v4u;
typedef __attribute__((ext_vector_type(2)))  unsigned int  v2u;

// Fragment viewed either as WMMA operand or as two 16B quads for b128 loads
union AB16 {
  v16bf v;
  v4u   q[2];
};

// ---- helpers -------------------------------------------------------------

__device__ inline unsigned short f2bf_u(float f) {
  // round-to-nearest-even fp32 -> bf16 (bit-level)
  unsigned u = __builtin_bit_cast(unsigned, f);
  unsigned r = u + 0x7FFFu + ((u >> 16) & 1u);
  return (unsigned short)(r >> 16);
}
__device__ inline unsigned pack2(float lo, float hi) {
  return (unsigned)f2bf_u(lo) | ((unsigned)f2bf_u(hi) << 16);
}
__device__ inline float wave_max(float v) {
  #pragma unroll
  for (int off = 16; off > 0; off >>= 1) v = fmaxf(v, __shfl_xor(v, off, 32));
  return v;
}
__device__ inline float wave_sum(float v) {
  #pragma unroll
  for (int off = 16; off > 0; off >>= 1) v += __shfl_xor(v, off, 32);
  return v;
}
__device__ inline float lrelu02(float e) { return e > 0.0f ? e : 0.2f * e; }
__device__ inline float elu1(float z)    { return z > 0.0f ? z : (__expf(z) - 1.0f); }

// ---- elementwise utility kernels ----------------------------------------

__global__ void k_zero(float* p, int n) {
  int i = blockIdx.x * blockDim.x + threadIdx.x;
  if (i < n) p[i] = 0.0f;
}

__global__ void k_init_relu(const float* __restrict__ in, float* __restrict__ x, int n) {
  int i = blockIdx.x * blockDim.x + threadIdx.x;
  if (i < n) x[i] = fmaxf(in[i], 0.0f);
}

// WT[d*D + k] = bf16( W[k*D + d] )   (transpose so WMMA B-fragments are K-contiguous)
__global__ void k_cvt_WT(const float* __restrict__ W, unsigned short* __restrict__ WT) {
  int idx = blockIdx.x * blockDim.x + threadIdx.x;  // 0 .. D*D-1
  int d = idx / D_, k = idx % D_;
  WT[d * D_ + k] = f2bf_u(W[k * D_ + d]);
}

// ---- ah1/ah2 + bf16(x): one wave per (b,i) row ---------------------------

__global__ void __launch_bounds__(128)
k_ah(const float* __restrict__ x, const float* __restrict__ w1, const float* __restrict__ b1,
     const float* __restrict__ w2, const float* __restrict__ b2,
     float* __restrict__ ah1, float* __restrict__ ah2, unsigned short* __restrict__ xbf) {
  int wave = (blockIdx.x * blockDim.x + threadIdx.x) >> 5;   // flat (b*N + i)
  int lane = threadIdx.x & 31;
  const float* row = x + (size_t)wave * D_;
  unsigned short* rowbf = xbf + (size_t)wave * D_;
  int d = lane * 4;
  float4 v  = *(const float4*)(row + d);
  float4 q1 = *(const float4*)(w1 + d);
  float4 q2 = *(const float4*)(w2 + d);
  v2u packed = { pack2(v.x, v.y), pack2(v.z, v.w) };
  *(v2u*)(rowbf + d) = packed;                    // b64 store of 4 bf16
  float s1 = v.x * q1.x + v.y * q1.y + v.z * q1.z + v.w * q1.w;
  float s2 = v.x * q2.x + v.y * q2.y + v.z * q2.z + v.w * q2.w;
  s1 = wave_sum(s1);
  s2 = wave_sum(s2);
  if (lane == 0) {
    ah1[wave] = s1 + b1[0];
    ah2[wave] = s2 + b2[0];
  }
}

// ---- xW = x @ W via bf16 WMMA: one wave per 16x16 tile -------------------
// Output written TRANSPOSED: xWT[b][d][i]  (K-major for the next matmul's B side)

__global__ void __launch_bounds__(128)
k_xw(const unsigned short* __restrict__ xbf, const unsigned short* __restrict__ WT,
     unsigned short* __restrict__ xWT) {
  int w    = (blockIdx.x * blockDim.x + threadIdx.x) >> 5;
  int lane = threadIdx.x & 31;
  int b    = w / (NT * DT);
  int rem  = w % (NT * DT);
  int it   = rem / DT, dt = rem % DT;
  int M    = lane & 15, half = lane >> 4;

  // A: row (it*16 + M) of x_bf16, K-contiguous.  B: column (dt*16 + M) of W, K-contiguous in WT.
  const unsigned short* xrow = xbf + ((size_t)b * N_ + (size_t)it * 16 + M) * D_;
  const unsigned short* wcol = WT + (size_t)(dt * 16 + M) * D_;
  v8f c = {};
  for (int kk = 0; kk < D_; kk += 32) {
    AB16 a, bm;
    a.q[0]  = *(const v4u*)(xrow + kk + half * 8);        // e=0..7  -> K = half*8 + e
    a.q[1]  = *(const v4u*)(xrow + kk + 16 + half * 8);   // e=8..15 -> K = 16 + half*8 + (e-8)
    bm.q[0] = *(const v4u*)(wcol + kk + half * 8);        // e=0..7  -> K = half*8 + e
    bm.q[1] = *(const v4u*)(wcol + kk + 16 + half * 8);   // e=8..15 -> K = 16 + half*8 + (e-8)
    c = __builtin_amdgcn_wmma_f32_16x16x32_bf16(false, a.v, false, bm.v, (short)0, c, false, false);
  }
  // C/D layout: VGPR r -> row M = r + half*8, col = lane&15.
  // In transposed storage the 8 rows are CONTIGUOUS for a fixed column -> one b128 store.
  unsigned short* dst = xWT + (size_t)(b * D_ + dt * 16 + M) * N_ + (size_t)it * 16 + half * 8;
  v4u o = { pack2(c[0], c[1]), pack2(c[2], c[3]), pack2(c[4], c[5]), pack2(c[6], c[7]) };
  *(v4u*)dst = o;
}

// ---- softmax pass 1: edge-only rowmax & 1/sumexp, one wave per (b,i) -----

__global__ void __launch_bounds__(128)
k_pass1(const float* __restrict__ g, const float* __restrict__ ah1, const float* __restrict__ ah2,
        float* __restrict__ rmax, float* __restrict__ rinv, int dir) {
  int w    = (blockIdx.x * blockDim.x + threadIdx.x) >> 5;
  int lane = threadIdx.x & 31;
  int b = w / N_, i = w % N_;
  float a1 = ah1[w];
  const float* ah2b = ah2 + (size_t)b * N_;
  float m = -3.0e38f;
  for (int j0 = 0; j0 < N_; j0 += 32) {
    int j = j0 + lane;
    float gv = dir ? g[((size_t)b * N_ + j) * N_ + i] : g[((size_t)b * N_ + i) * N_ + j];
    if (gv > 0.0f) m = fmaxf(m, lrelu02(a1 + ah2b[j]));
  }
  m = wave_max(m);
  float s = 0.0f;
  if (m > -1.0e38f) {   // uniform across wave
    for (int j0 = 0; j0 < N_; j0 += 32) {
      int j = j0 + lane;
      float gv = dir ? g[((size_t)b * N_ + j) * N_ + i] : g[((size_t)b * N_ + i) * N_ + j];
      if (gv > 0.0f) s += __expf(lrelu02(a1 + ah2b[j]) - m);
    }
    s = wave_sum(s);
  }
  if (lane == 0) {
    rmax[w] = m;
    rinv[w] = (s > 0.0f) ? 1.0f / s : 0.0f;
  }
}

// ---- trans = a @ xW + fused elu(x+trans): 1 block per 16-row tile --------

__global__ void __launch_bounds__(128)
k_attmm(const float* __restrict__ xcur, const unsigned short* __restrict__ xWT,
        const float* __restrict__ g, const float* __restrict__ ah1, const float* __restrict__ ah2,
        const float* __restrict__ rmax, const float* __restrict__ rinv,
        float* __restrict__ xnext, int dir) {
  __shared__ alignas(16) unsigned short sA[16 * 32];
  __shared__ float s_a1[16], s_m[16], s_inv[16];

  int tid  = threadIdx.x;
  int b    = blockIdx.x / NT;
  int it   = blockIdx.x % NT;
  int lane = tid & 31, wv = tid >> 5;
  int M    = lane & 15, half = lane >> 4;

  if (tid < 16) {
    int i = it * 16 + tid;
    s_a1[tid]  = ah1[(size_t)b * N_ + i];
    s_m[tid]   = rmax[(size_t)b * N_ + i];
    s_inv[tid] = rinv[(size_t)b * N_ + i];
  }
  const float* ah2b = ah2 + (size_t)b * N_;
  int dt0 = wv * 2, dt1 = wv * 2 + 1;
  // Columns owned by this lane for the two D-tiles (K-contiguous in xWT)
  const unsigned short* xwc0 = xWT + (size_t)(b * D_ + dt0 * 16 + M) * N_;
  const unsigned short* xwc1 = xWT + (size_t)(b * D_ + dt1 * 16 + M) * N_;
  v8f c0 = {}, c1 = {};

  int idx0 = tid * 4;                    // each thread fills 4 consecutive sA entries
  int mrow = idx0 >> 5, jj0 = idx0 & 31; // same row for all 4 (4-aligned inside 32)

  for (int j0 = 0; j0 < N_; j0 += 32) {
    __syncthreads();                     // also covers s_a1/s_m/s_inv init
    {
      int i = it * 16 + mrow, j = j0 + jj0;
      float gvv[4];
      if (dir == 0) {
        float4 g4 = *(const float4*)(g + ((size_t)b * N_ + i) * N_ + j);
        gvv[0] = g4.x; gvv[1] = g4.y; gvv[2] = g4.z; gvv[3] = g4.w;
      } else {
        #pragma unroll
        for (int q = 0; q < 4; q++) gvv[q] = g[((size_t)b * N_ + j + q) * N_ + i];
      }
      float4 h4 = *(const float4*)(ah2b + j);
      float hv[4] = { h4.x, h4.y, h4.z, h4.w };
      float av[4];
      #pragma unroll
      for (int q = 0; q < 4; q++) {
        float a = 0.0f;
        if (gvv[q] > 0.0f)
          a = __expf(lrelu02(s_a1[mrow] + hv[q]) - s_m[mrow]) * s_inv[mrow];
        av[q] = a;
      }
      v2u packed = { pack2(av[0], av[1]), pack2(av[2], av[3]) };
      *(v2u*)(sA + idx0) = packed;       // b64 LDS store
    }
    __syncthreads();

    AB16 a, b0, b1;
    const unsigned short* sArow = sA + M * 32;
    a.q[0]  = *(const v4u*)(sArow + half * 8);          // ds_load_b128
    a.q[1]  = *(const v4u*)(sArow + 16 + half * 8);
    b0.q[0] = *(const v4u*)(xwc0 + j0 + half * 8);      // global b128, K-contiguous
    b0.q[1] = *(const v4u*)(xwc0 + j0 + 16 + half * 8);
    b1.q[0] = *(const v4u*)(xwc1 + j0 + half * 8);
    b1.q[1] = *(const v4u*)(xwc1 + j0 + 16 + half * 8);
    c0 = __builtin_amdgcn_wmma_f32_16x16x32_bf16(false, a.v, false, b0.v, (short)0, c0, false, false);
    c1 = __builtin_amdgcn_wmma_f32_16x16x32_bf16(false, a.v, false, b1.v, (short)0, c1, false, false);
  }

  // epilogue: x_next = elu(x + trans)
  #pragma unroll
  for (int r = 0; r < 8; r++) {
    int i = it * 16 + r + half * 8;
    size_t base = ((size_t)b * N_ + i) * D_;
    int d0 = dt0 * 16 + M, d1 = dt1 * 16 + M;
    xnext[base + d0] = elu1(xcur[base + d0] + c0[r]);
    xnext[base + d1] = elu1(xcur[base + d1] + c1[r]);
  }
}

// ---- mid[b,d] += sum_i x[b,i,d] ------------------------------------------

__global__ void __launch_bounds__(128)
k_accum(const float* __restrict__ x, float* __restrict__ mid) {
  int b = blockIdx.x / 8;
  int chunk = blockIdx.x % 8;
  int d = threadIdx.x;                       // 0..127
  float s = 0.0f;
  for (int i = chunk * 256; i < chunk * 256 + 256; i++)
    s += x[((size_t)b * N_ + i) * D_ + d];
  atomicAdd(&mid[b * D_ + d], s);
}

// ---- out = mid @ Wout + bout ---------------------------------------------

__global__ void k_final(const float* __restrict__ mid, const float* __restrict__ Wout,
                        const float* __restrict__ bout, float* __restrict__ out) {
  int b = blockIdx.x;
  int o = threadIdx.x;                       // 0..63
  float s = bout[o];
  for (int d = 0; d < D_; d++) s += mid[b * D_ + d] * Wout[d * OUT_ + o];
  out[b * OUT_ + o] = s;
}

// ---- host launch ---------------------------------------------------------

extern "C" void kernel_launch(void* const* d_in, const int* in_sizes, int n_in,
                              void* d_out, int out_size, void* d_ws, size_t ws_size,
                              hipStream_t stream) {
  (void)in_sizes; (void)n_in; (void)out_size; (void)ws_size;
  const float* INPUT = (const float*)d_in[0];
  const float* CFG   = (const float*)d_in[1];
  /* d_in[2] = LFG, unused by forward */
  const float* w1   = (const float*)d_in[3];
  const float* b1   = (const float*)d_in[4];
  const float* w2   = (const float*)d_in[5];
  const float* b2   = (const float*)d_in[6];
  const float* W    = (const float*)d_in[7];
  const float* Wout = (const float*)d_in[8];
  const float* bout = (const float*)d_in[9];
  float* out = (float*)d_out;

  char* ws = (char*)d_ws;
  size_t off = 0;
  auto alloc = [&](size_t bytes) -> void* {
    void* p = ws + off;
    off = (off + bytes + 255) & ~(size_t)255;
    return p;
  };
  float*          x_a  = (float*)alloc((size_t)B_ * N_ * D_ * 4);
  float*          x_b  = (float*)alloc((size_t)B_ * N_ * D_ * 4);
  unsigned short* xbf  = (unsigned short*)alloc((size_t)B_ * N_ * D_ * 2);
  unsigned short* xWT  = (unsigned short*)alloc((size_t)B_ * N_ * D_ * 2);  // transposed [b][d][i]
  unsigned short* WT   = (unsigned short*)alloc((size_t)D_ * D_ * 2);       // transposed [d][k]
  float*          ah1  = (float*)alloc((size_t)B_ * N_ * 4);
  float*          ah2  = (float*)alloc((size_t)B_ * N_ * 4);
  float*          rmax = (float*)alloc((size_t)B_ * N_ * 4);
  float*          rinv = (float*)alloc((size_t)B_ * N_ * 4);
  float*          mid  = (float*)alloc((size_t)B_ * D_ * 4);

  k_zero<<<1, 512, 0, stream>>>(mid, B_ * D_);
  k_cvt_WT<<<(D_ * D_ + 255) / 256, 256, 0, stream>>>(W, WT);

  for (int dir = 0; dir < 2; dir++) {
    int n = B_ * N_ * D_;
    k_init_relu<<<(n + 255) / 256, 256, 0, stream>>>(INPUT, x_a, n);
    float* cur = x_a;
    float* nxt = x_b;
    for (int t = 0; t < 3; t++) {
      k_ah   <<<(B_ * N_ * 32) / 128, 128, 0, stream>>>(cur, w1, b1, w2, b2, ah1, ah2, xbf);
      k_xw   <<<(B_ * NT * DT * 32) / 128, 128, 0, stream>>>(xbf, WT, xWT);
      k_pass1<<<(B_ * N_ * 32) / 128, 128, 0, stream>>>(CFG, ah1, ah2, rmax, rinv, dir);
      k_attmm<<<B_ * NT, 128, 0, stream>>>(cur, xWT, CFG, ah1, ah2, rmax, rinv, nxt, dir);
      float* tmp = cur; cur = nxt; nxt = tmp;
    }
    k_accum<<<B_ * 8, 128, 0, stream>>>(cur, mid);
  }
  k_final<<<B_, OUT_, 0, stream>>>(mid, Wout, bout, out);
}
